// Head_19250043420980
// MI455X (gfx1250) — matrix operand order repaired and checked by reading, not compiled, not run on hardware
//
#include <hip/hip_runtime.h>
#include <hip/hip_bf16.h>

// Problem constants (match reference setup_inputs)
#define BB 8
#define TT 2048
#define CC 1024
#define HH 64

typedef __attribute__((ext_vector_type(16))) _Float16 v16h;
typedef __attribute__((ext_vector_type(8)))  _Float16 h8;
typedef __attribute__((ext_vector_type(8)))  float    v8f;
typedef __attribute__((ext_vector_type(4)))  float    f4;

// ---- WMMA operand loaders (f16, contiguous-K order, same for A and B) -----------------
// A 16x32: lane l (m=l&15, hi=l>>4) holds contiguous K = hi*16 .. hi*16+15.
// Caller passes p already offset by (kbase + hi*16); chunks at +0 and +8.
static __device__ __forceinline__ v16h ldA16(const _Float16* p) {
    h8 c0 = *(const h8*)p;
    h8 c1 = *(const h8*)(p + 8);
    v16h r;
#pragma unroll
    for (int i = 0; i < 8; ++i) { r[i] = c0[i]; r[i + 8] = c1[i]; }
    return r;
}
// B 32x16: lane l (n=l&15, hi=l>>4) holds contiguous K = hi*16 .. hi*16+15.
// Caller passes p already offset by (kbase + hi*16); chunks at +0 and +8.
static __device__ __forceinline__ v16h ldB16(const _Float16* p) {
    h8 c0 = *(const h8*)p;
    h8 c1 = *(const h8*)(p + 8);
    v16h r;
#pragma unroll
    for (int i = 0; i < 8; ++i) { r[i] = c0[i]; r[i + 8] = c1[i]; }
    return r;
}
static __device__ __forceinline__ v8f wmma_f16(v16h a, v16h b, v8f c) {
    return __builtin_amdgcn_wmma_f32_16x16x32_f16(false, a, false, b, (short)0, c,
                                                  false, false);
}

// ---- Kernel 0a: weights fp32 [C][H] -> f16 wt[mat][h][c] --------------------
__global__ void head_wtrans(const float* __restrict__ Wq,
                            const float* __restrict__ Wk,
                            const float* __restrict__ Wv,
                            _Float16* __restrict__ wt) {
    int idx = blockIdx.x * blockDim.x + threadIdx.x;
    const int total = 3 * HH * CC;
    if (idx >= total) return;
    int mat = idx / (HH * CC);
    int r   = idx % (HH * CC);
    int h   = r / CC;
    int c   = r % CC;
    const float* W = (mat == 0) ? Wq : (mat == 1) ? Wk : Wv;
    wt[idx] = (_Float16)W[c * HH + h];
}

// ---- Kernel 0b: x fp32 -> f16 (vectorized by 8) -----------------------------
__global__ void head_xcvt(const float* __restrict__ x, _Float16* __restrict__ xh) {
    int i = blockIdx.x * blockDim.x + threadIdx.x;   // [0, B*T*C/8)
    const float* p = x + (size_t)i * 8;
    f4 a = *(const f4*)(p);
    f4 b = *(const f4*)(p + 4);
    h8 o;
#pragma unroll
    for (int j = 0; j < 4; ++j) { o[j] = (_Float16)a[j]; o[j + 4] = (_Float16)b[j]; }
    *(h8*)(xh + (size_t)i * 8) = o;
}

// ---- Kernel 1: projections. One wave = (b, t-tile, matrix), all 4 h-tiles ---
// q,k row-major [T][H]; v transposed [H][T]. A operand loaded once per K-step
// and reused by 4 WMMAs (one per 16-col h tile).
__global__ void head_proj(const float* __restrict__ x,
                          const _Float16* __restrict__ xh,
                          const _Float16* __restrict__ wt,
                          _Float16* __restrict__ qh,
                          _Float16* __restrict__ kh,
                          _Float16* __restrict__ vt,
                          int use_xh) {
    int gtid = blockIdx.x * blockDim.x + threadIdx.x;
    int wave = gtid >> 5;                      // [0, B*3*(T/16)) = 3072
    int lane = threadIdx.x & 31;
    int b    = wave / (3 * (TT / 16));
    int rem  = wave % (3 * (TT / 16));
    int mat  = rem / (TT / 16);
    int t0   = (rem % (TT / 16)) * 16;
    int m  = lane & 15;
    int hi = lane >> 4;

    const _Float16* w0 = wt + (size_t)mat * HH * CC + (size_t)m * CC; // + j*16*CC
    v8f a0 = {}, a1 = {}, a2 = {}, a3 = {};

    auto mmstep = [&](v16h a, int c) {
        a0 = wmma_f16(a, ldB16(w0 + (size_t)(0)       * CC + c + hi * 16), a0);
        a1 = wmma_f16(a, ldB16(w0 + (size_t)(16)      * CC + c + hi * 16), a1);
        a2 = wmma_f16(a, ldB16(w0 + (size_t)(32)      * CC + c + hi * 16), a2);
        a3 = wmma_f16(a, ldB16(w0 + (size_t)(48)      * CC + c + hi * 16), a3);
    };

    if (use_xh) {
        const _Float16* xr = xh + ((size_t)b * TT + (t0 + m)) * CC;
        for (int c = 0; c < CC; c += 32) mmstep(ldA16(xr + c + hi * 16), c);
    } else {
        const float* xr = x + ((size_t)b * TT + (t0 + m)) * CC;
        for (int c = 0; c < CC; c += 32) {
            const float* p0 = xr + c + hi * 16;
            f4 x0 = *(const f4*)(p0);
            f4 x1 = *(const f4*)(p0 + 4);
            f4 x2 = *(const f4*)(p0 + 8);
            f4 x3 = *(const f4*)(p0 + 12);
            v16h a;
#pragma unroll
            for (int i = 0; i < 4; ++i) {
                a[i]      = (_Float16)x0[i];
                a[i + 4]  = (_Float16)x1[i];
                a[i + 8]  = (_Float16)x2[i];
                a[i + 12] = (_Float16)x3[i];
            }
            mmstep(a, c);
        }
    }

    // D element (t0 + r + 8*hi, j*16 + m)
    v8f accs[4] = {a0, a1, a2, a3};
    if (mat == 2) {
        // store v transposed: vt[b][h][t]; r-consecutive => t-contiguous h8 store
#pragma unroll
        for (int j = 0; j < 4; ++j) {
            h8 vrow;
#pragma unroll
            for (int r = 0; r < 8; ++r) vrow[r] = (_Float16)accs[j][r];
            _Float16* dst = vt + (size_t)b * HH * TT + (size_t)(j * 16 + m) * TT
                               + (t0 + 8 * hi);
            *(h8*)dst = vrow;
        }
    } else {
        _Float16* dst = (mat == 0) ? qh : kh;
#pragma unroll
        for (int j = 0; j < 4; ++j) {
#pragma unroll
            for (int r = 0; r < 8; ++r) {
                dst[((size_t)b * TT + (t0 + r + 8 * hi)) * HH + j * 16 + m] =
                    (_Float16)accs[j][r];
            }
        }
    }
}

// ---- Kernel 2: per-column (axis=1 softmax) max & exp-sum --------------------
// One wave per (batch, 16-column block); streams row tiles t >= s0.
__global__ void head_colstats(const _Float16* __restrict__ qh,
                              const _Float16* __restrict__ kh,
                              float* __restrict__ colm,
                              float* __restrict__ coll) {
    int wid = blockIdx.x;                 // [0, B * T/16)
    int b   = wid / (TT / 16);
    int s0  = (wid % (TT / 16)) * 16;
    int lane = threadIdx.x & 31;
    int m  = lane & 15;
    int hi = lane >> 4;

    // Preload B operand: k columns s0+m (contiguous over h)
    const _Float16* kcol = kh + ((size_t)b * TT + (s0 + m)) * HH;
    v16h bk0 = ldB16(kcol + 0  + hi * 16);
    v16h bk1 = ldB16(kcol + 32 + hi * 16);

    float mstat = -3.0e38f;
    float lstat = 0.0f;
    int scol = s0 + m;

    for (int t0 = s0; t0 < TT; t0 += 16) {
        const _Float16* qrow = qh + ((size_t)b * TT + (t0 + m)) * HH;
        v8f acc = {};
        acc = wmma_f16(ldA16(qrow + 0  + hi * 16), bk0, acc);
        acc = wmma_f16(ldA16(qrow + 32 + hi * 16), bk1, acc);

        float tmax = -3.0e38f;
#pragma unroll
        for (int r = 0; r < 8; ++r) {
            int t = t0 + r + 8 * hi;
            if (scol <= t) tmax = fmaxf(tmax, acc[r]);
        }
        tmax = fmaxf(tmax, __shfl_xor(tmax, 16));
        float mnew = fmaxf(mstat, tmax);
        float tsum = 0.0f;
#pragma unroll
        for (int r = 0; r < 8; ++r) {
            int t = t0 + r + 8 * hi;
            if (scol <= t) tsum += __expf(acc[r] - mnew);
        }
        tsum += __shfl_xor(tsum, 16);
        lstat = lstat * __expf(mstat - mnew) + tsum;
        mstat = mnew;
    }
    if (lane < 16) {
        colm[(size_t)b * TT + s0 + lane] = mstat;
        coll[(size_t)b * TT + s0 + lane] = lstat;
    }
}

// ---- Kernel 3: out[t,h] = sum_s exp(S[t,s]-m[s])/l[s] * v[s,h] --------------
// One wave per (batch, 16-row q tile); loops s-blocks of 32 (causal bound).
__global__ void head_attnout(const _Float16* __restrict__ qh,
                             const _Float16* __restrict__ kh,
                             const _Float16* __restrict__ vt,
                             const float* __restrict__ colm,
                             const float* __restrict__ coll,
                             float* __restrict__ out) {
    __shared__ __align__(16) _Float16 pbuf[16 * 32];
    int wid = blockIdx.x;
    int b   = wid / (TT / 16);
    int t0  = (wid % (TT / 16)) * 16;
    int lane = threadIdx.x & 31;
    int m  = lane & 15;
    int hi = lane >> 4;

    const _Float16* qrow = qh + ((size_t)b * TT + (t0 + m)) * HH;
    v16h aq0 = ldA16(qrow + 0  + hi * 16);
    v16h aq1 = ldA16(qrow + 32 + hi * 16);

    const _Float16* vb = vt + (size_t)b * HH * TT;
    const float* mrow = colm + (size_t)b * TT;
    const float* lrow = coll + (size_t)b * TT;

    v8f o0 = {}, o1 = {}, o2 = {}, o3 = {};

    for (int sb = 0; sb < t0 + 16; sb += 32) {
        // scores for s in [sb, sb+32)
        const _Float16* kc0 = kh + ((size_t)b * TT + (sb + m)) * HH;
        const _Float16* kc1 = kc0 + 16 * HH;
        v8f sA = {}, sBv = {};
        sA  = wmma_f16(aq0, ldB16(kc0 + 0  + hi * 16), sA);
        sA  = wmma_f16(aq1, ldB16(kc0 + 32 + hi * 16), sA);
        sBv = wmma_f16(aq0, ldB16(kc1 + 0  + hi * 16), sBv);
        sBv = wmma_f16(aq1, ldB16(kc1 + 32 + hi * 16), sBv);

        int scol0 = sb + m;
        int scol1 = sb + 16 + m;
        float mv0 = mrow[scol0], rl0 = 1.0f / lrow[scol0];
        float mv1 = mrow[scol1], rl1 = 1.0f / lrow[scol1];

        // p = exp(S - m[s]) / l[s], causal mask; stage to LDS [t(16)][s(32)]
#pragma unroll
        for (int r = 0; r < 8; ++r) {
            int t = t0 + r + 8 * hi;
            float p0 = (scol0 <= t) ? __expf(sA[r]  - mv0) * rl0 : 0.0f;
            float p1 = (scol1 <= t) ? __expf(sBv[r] - mv1) * rl1 : 0.0f;
            pbuf[(r + 8 * hi) * 32 + m]      = (_Float16)p0;
            pbuf[(r + 8 * hi) * 32 + 16 + m] = (_Float16)p1;
        }
        __syncthreads();   // single-wave workgroup: lowers to S_NOP, LDS in-order

        // p (16x32) x v (32x64): A from LDS, B from transposed v (contiguous s)
        v16h ap = ldA16(pbuf + m * 32 + hi * 16);
        o0 = wmma_f16(ap, ldB16(vb + (size_t)(0  + m) * TT + sb + hi * 16), o0);
        o1 = wmma_f16(ap, ldB16(vb + (size_t)(16 + m) * TT + sb + hi * 16), o1);
        o2 = wmma_f16(ap, ldB16(vb + (size_t)(32 + m) * TT + sb + hi * 16), o2);
        o3 = wmma_f16(ap, ldB16(vb + (size_t)(48 + m) * TT + sb + hi * 16), o3);
        __syncthreads();
    }

#pragma unroll
    for (int r = 0; r < 8; ++r) {
        size_t row = ((size_t)b * TT + t0 + r + 8 * hi) * HH;
        out[row + 0  + m] = o0[r];
        out[row + 16 + m] = o1[r];
        out[row + 32 + m] = o2[r];
        out[row + 48 + m] = o3[r];
    }
}

// ---- Launcher ---------------------------------------------------------------
extern "C" void kernel_launch(void* const* d_in, const int* in_sizes, int n_in,
                              void* d_out, int out_size, void* d_ws, size_t ws_size,
                              hipStream_t stream) {
    const float* x  = (const float*)d_in[0];   // [B,T,C]
    const float* Wq = (const float*)d_in[1];   // [C,H]
    const float* Wk = (const float*)d_in[2];
    const float* Wv = (const float*)d_in[3];
    float* out = (float*)d_out;                // [B,T,H]

    // Workspace layout (base 6.5 MB; optional +32 MB f16 copy of x)
    char* ws = (char*)d_ws;
    _Float16* wt = (_Float16*)(ws);                       // 3*64*1024 f16 = 0x60000
    _Float16* qh = (_Float16*)(ws + 0x060000);            // B*T*H f16 = 2 MB
    _Float16* kh = (_Float16*)(ws + 0x260000);            // 2 MB
    _Float16* vt = (_Float16*)(ws + 0x460000);            // 2 MB (layout [b][h][t])
    float*  colm = (float*)   (ws + 0x660000);            // B*T f32 = 64 KB
    float*  coll = (float*)   (ws + 0x670000);            // 64 KB  (end 0x680000)
    const size_t xh_off   = 0x680000;
    const size_t xh_bytes = (size_t)BB * TT * CC * sizeof(_Float16);  // 32 MB
    int use_xh = (ws_size >= xh_off + xh_bytes) ? 1 : 0;  // uniform, deterministic
    _Float16* xh = (_Float16*)(ws + xh_off);
    (void)in_sizes; (void)n_in; (void)out_size;

    // 0a) transpose weights to f16
    {
        int total = 3 * HH * CC;
        head_wtrans<<<(total + 255) / 256, 256, 0, stream>>>(Wq, Wk, Wv, wt);
    }
    // 0b) pre-convert x to f16 if workspace allows (halves GEMM A bandwidth)
    if (use_xh) {
        int total8 = BB * TT * CC / 8;                    // 2,097,152
        head_xcvt<<<total8 / 256, 256, 0, stream>>>(x, xh);
    }
    // 1) projections: B*3*(T/16) = 3072 waves (4 h-tiles each), 8 waves/block
    head_proj<<<3072 / 8, 256, 0, stream>>>(x, xh, wt, qh, kh, vt, use_xh);
    // 2) column-softmax stats: one wave per (b, 16 cols)
    head_colstats<<<BB * (TT / 16), 32, 0, stream>>>(qh, kh, colm, coll);
    // 3) attention output: one wave per (b, 16 rows)
    head_attnout<<<BB * (TT / 16), 32, 0, stream>>>(qh, kh, vt, colm, coll, out);
}
